// MIT_20057497272559
// MI455X (gfx1250) — compile-verified, matrix operand order checked by reading, not run on hardware
//
#include <hip/hip_runtime.h>
#include <hip/hip_bf16.h>
#include <math.h>
#include <stdint.h>

// ---------------- problem constants ----------------
constexpr int B = 4, N = 60000, S = 1024, V = 8, D = 256, H = 4, DEPTH = 4, C = 20;
constexpr int DH = D / H;        // 64
constexpr int MH = 4 * D;        // 1024
constexpr int TV = C + V;        // 28  view tokens
constexpr int TX = C + S;        // 1044 voxel tokens
#define SCALE_ATT 0.125f         // 1/sqrt(64)

typedef __attribute__((ext_vector_type(16))) __bf16 v16bf;
typedef __attribute__((ext_vector_type(8)))  __bf16 v8bf;
typedef __attribute__((ext_vector_type(4)))  __bf16 v4bf;
typedef __attribute__((ext_vector_type(8)))  float  v8f;

// ============================================================
// bf16 WMMA GEMM:  C = act( scale * (A @ W(^T)) + bias + Resid )
//   A : M x K f32 (lda)
//   W : WB==1 -> pre-converted bf16, N x K row-major (ldw = K)
//       WB==0, WT==1 -> f32 N x K (ldw)        (A @ W^T : Q@K^T)
//       WB==0, WT==0 -> f32 K x N (ldw)        (P@V)
//   batched over gridDim.z: z -> (zb = z/Hn, zh = z%Hn) offsets
// Block: 128 threads = 4 wave32; tile 64x64, K step 32; wave = 32x32 = 2x2 wmma.
// Double-buffered LDS, one barrier per K-step. A-tiles (f32->bf16) prefetch
// through registers; bf16 weight tiles go straight global->LDS with
// GLOBAL_LOAD_ASYNC_TO_LDS_B128 (ASYNCcnt), fenced before the publish barrier.
// ============================================================
template<int ACT, int WT, int WB>
__global__ __launch_bounds__(128)
void gemm_k(const float* __restrict__ A, const void* __restrict__ Wvoid,
            const float* __restrict__ bias, const float* __restrict__ Resid,
            float* __restrict__ Cout,
            int M, int K, int Ncols, int lda, int ldw, int ldc, float scale,
            long long aOffB, long long aOffH, long long wOffB, long long wOffH,
            long long cOffB, long long cOffH, int Hn)
{
    __shared__ __attribute__((aligned(16))) __bf16 As[2][64][40];
    __shared__ __attribute__((aligned(16))) __bf16 Bs[2][64][40];

    const float*  Wf = (const float*) Wvoid;
    const __bf16* Wb = (const __bf16*)Wvoid;

    int z = blockIdx.z, zb = z / Hn, zh = z % Hn;
    A += (long long)zb * aOffB + (long long)zh * aOffH;
    long long woff = (long long)zb * wOffB + (long long)zh * wOffH;
    long long coff = (long long)zb * cOffB + (long long)zh * cOffH;
    Cout += coff;
    const float* Rp = Resid ? Resid + coff : nullptr;

    int tid = threadIdx.x, lane = tid & 31, wv = tid >> 5;
    int wm = wv >> 1, wn = wv & 1;
    int row0 = blockIdx.y * 64, col0 = blockIdx.x * 64;

    const bool aFull = (row0 + 64 <= M);
    const bool bFull = (col0 + 64 <= Ncols);
    const bool aVec  = ((lda & 3) == 0);
    const bool wVec  = ((ldw & 3) == 0);
    const bool wAsync8 = WB && ((ldw & 7) == 0);   // 16B-aligned weight rows

    // prefetch registers (one 64x32 tile each, 4 groups of 4 per thread)
    float4 aQ[4];
    float4 wQf[4];
    v4bf   wQb[4];

    // ---------------- load phase: pure global loads ----------------
    auto loadA = [&](int k0) {
        const bool kFull = (k0 + 32 <= K);
        if (aFull && kFull && aVec) {
            #pragma unroll
            for (int i = 0; i < 4; ++i) {
                int e = (i * 128 + tid) << 2;
                int m = e >> 5, kk = e & 31;
                aQ[i] = *(const float4*)(A + (long long)(row0 + m) * lda + (k0 + kk));
            }
        } else {
            #pragma unroll
            for (int i = 0; i < 4; ++i) {
                int e = (i * 128 + tid) << 2;
                int m = e >> 5, kk = e & 31;
                int gr = row0 + m, gk = k0 + kk;
                bool okr = (gr < M);
                long long base = (long long)gr * lda + gk;
                aQ[i].x = A[(okr && gk + 0 < K) ? base + 0 : 0];
                aQ[i].y = A[(okr && gk + 1 < K) ? base + 1 : 0];
                aQ[i].z = A[(okr && gk + 2 < K) ? base + 2 : 0];
                aQ[i].w = A[(okr && gk + 3 < K) ? base + 3 : 0];
            }
        }
    };
    auto loadB = [&](int k0) {
        const bool kFull = (k0 + 32 <= K);
        if (WB) {
            if (bFull && kFull && wVec) {
                #pragma unroll
                for (int i = 0; i < 4; ++i) {
                    int e = (i * 128 + tid) << 2;
                    int n = e >> 5, kk = e & 31;
                    wQb[i] = *(const v4bf*)(Wb + woff + (long long)(col0 + n) * ldw + (k0 + kk));
                }
            } else {
                #pragma unroll
                for (int i = 0; i < 4; ++i) {
                    int e = (i * 128 + tid) << 2;
                    int n = e >> 5, kk = e & 31;
                    int gn = col0 + n, gk = k0 + kk;
                    bool okn = (gn < Ncols);
                    long long base = woff + (long long)gn * ldw + gk;
                    #pragma unroll
                    for (int j = 0; j < 4; ++j)
                        wQb[i][j] = Wb[(okn && gk + j < K) ? base + j : 0];
                }
            }
        } else if (WT) {
            if (bFull && kFull && wVec) {
                #pragma unroll
                for (int i = 0; i < 4; ++i) {
                    int e = (i * 128 + tid) << 2;
                    int n = e >> 5, kk = e & 31;
                    wQf[i] = *(const float4*)(Wf + woff + (long long)(col0 + n) * ldw + (k0 + kk));
                }
            } else {
                #pragma unroll
                for (int i = 0; i < 4; ++i) {
                    int e = (i * 128 + tid) << 2;
                    int n = e >> 5, kk = e & 31;
                    int gn = col0 + n, gk = k0 + kk;
                    bool okn = (gn < Ncols);
                    long long base = woff + (long long)gn * ldw + gk;
                    wQf[i].x = Wf[(okn && gk + 0 < K) ? base + 0 : 0];
                    wQf[i].y = Wf[(okn && gk + 1 < K) ? base + 1 : 0];
                    wQf[i].z = Wf[(okn && gk + 2 < K) ? base + 2 : 0];
                    wQf[i].w = Wf[(okn && gk + 3 < K) ? base + 3 : 0];
                }
            }
        } else {      // f32 K x N, vector along n
            if (bFull && kFull && wVec) {
                #pragma unroll
                for (int i = 0; i < 4; ++i) {
                    int e = (i * 128 + tid) << 2;
                    int kk = e >> 6, n = e & 63;
                    wQf[i] = *(const float4*)(Wf + woff + (long long)(k0 + kk) * ldw + (col0 + n));
                }
            } else {
                #pragma unroll
                for (int i = 0; i < 4; ++i) {
                    int e = (i * 128 + tid) << 2;
                    int kk = e >> 6, n = e & 63;
                    int gn = col0 + n, gk = k0 + kk;
                    bool okk = (gk < K);
                    long long base = woff + (long long)gk * ldw + gn;
                    wQf[i].x = Wf[(okk && gn + 0 < Ncols) ? base + 0 : 0];
                    wQf[i].y = Wf[(okk && gn + 1 < Ncols) ? base + 1 : 0];
                    wQf[i].z = Wf[(okk && gn + 2 < Ncols) ? base + 2 : 0];
                    wQf[i].w = Wf[(okk && gn + 3 < Ncols) ? base + 3 : 0];
                }
            }
        }
    };

    // ---------------- store phase: zero-substitute + convert + LDS ----------------
    auto storeA = [&](int bsel, int k0) {
        const bool edge = !(aFull && (k0 + 32 <= K));
        #pragma unroll
        for (int i = 0; i < 4; ++i) {
            int e = (i * 128 + tid) << 2;
            int m = e >> 5, kk = e & 31;
            float4 v = aQ[i];
            if (edge) {
                int gr = row0 + m, gk = k0 + kk;
                bool okr = (gr < M);
                if (!(okr && gk + 0 < K)) v.x = 0.f;
                if (!(okr && gk + 1 < K)) v.y = 0.f;
                if (!(okr && gk + 2 < K)) v.z = 0.f;
                if (!(okr && gk + 3 < K)) v.w = 0.f;
            }
            v4bf o; o[0] = (__bf16)v.x; o[1] = (__bf16)v.y; o[2] = (__bf16)v.z; o[3] = (__bf16)v.w;
            *(v4bf*)&As[bsel][m][kk] = o;
        }
    };
    auto storeB = [&](int bsel, int k0) {
        const bool edge = !(bFull && (k0 + 32 <= K));
        if (WB) {
            #pragma unroll
            for (int i = 0; i < 4; ++i) {
                int e = (i * 128 + tid) << 2;
                int n = e >> 5, kk = e & 31;
                v4bf o = wQb[i];
                if (edge) {
                    int gn = col0 + n, gk = k0 + kk;
                    bool okn = (gn < Ncols);
                    #pragma unroll
                    for (int j = 0; j < 4; ++j)
                        if (!(okn && gk + j < K)) o[j] = (__bf16)0.0f;
                }
                *(v4bf*)&Bs[bsel][n][kk] = o;
            }
        } else if (WT) {
            #pragma unroll
            for (int i = 0; i < 4; ++i) {
                int e = (i * 128 + tid) << 2;
                int n = e >> 5, kk = e & 31;
                float4 v = wQf[i];
                if (edge) {
                    int gn = col0 + n, gk = k0 + kk;
                    bool okn = (gn < Ncols);
                    if (!(okn && gk + 0 < K)) v.x = 0.f;
                    if (!(okn && gk + 1 < K)) v.y = 0.f;
                    if (!(okn && gk + 2 < K)) v.z = 0.f;
                    if (!(okn && gk + 3 < K)) v.w = 0.f;
                }
                v4bf o; o[0] = (__bf16)v.x; o[1] = (__bf16)v.y; o[2] = (__bf16)v.z; o[3] = (__bf16)v.w;
                *(v4bf*)&Bs[bsel][n][kk] = o;
            }
        } else {
            #pragma unroll
            for (int i = 0; i < 4; ++i) {
                int e = (i * 128 + tid) << 2;
                int kk = e >> 6, n = e & 63;
                float4 v = wQf[i];
                if (edge) {
                    int gn = col0 + n, gk = k0 + kk;
                    bool okk = (gk < K);
                    if (!(okk && gn + 0 < Ncols)) v.x = 0.f;
                    if (!(okk && gn + 1 < Ncols)) v.y = 0.f;
                    if (!(okk && gn + 2 < Ncols)) v.z = 0.f;
                    if (!(okk && gn + 3 < Ncols)) v.w = 0.f;
                }
                Bs[bsel][n + 0][kk] = (__bf16)v.x;
                Bs[bsel][n + 1][kk] = (__bf16)v.y;
                Bs[bsel][n + 2][kk] = (__bf16)v.z;
                Bs[bsel][n + 3][kk] = (__bf16)v.w;
            }
        }
    };

    // ---- CDNA5 async global->LDS copy of a bf16 weight tile (no VGPR round-trip) ----
    auto asyncB = [&](int bsel, int k0) {
        #pragma unroll
        for (int i = 0; i < 4; ++i) {
            int e = i * 128 + tid;                // 512 chunks of 16B
            int n = e >> 2, kk = (e & 3) * 8;     // 8 bf16 per chunk
            unsigned long long ga = (unsigned long long)(uintptr_t)
                (Wb + woff + (long long)(col0 + n) * ldw + (k0 + kk));
            unsigned la = (unsigned)(uintptr_t)&Bs[bsel][n][kk];
            asm volatile("global_load_async_to_lds_b128 %0, %1, off"
                         :: "v"(la), "v"(ga) : "memory");
        }
    };

    v8f acc[2][2];
    #pragma unroll
    for (int i = 0; i < 2; ++i)
        #pragma unroll
        for (int j = 0; j < 2; ++j)
            #pragma unroll
            for (int r = 0; r < 8; ++r) acc[i][j][r] = 0.0f;

    const int nt = (K + 31) / 32;
    // prologue: stage tile 0
    const bool fast0 = wAsync8 && bFull && (32 <= K);
    loadA(0);
    if (!fast0) loadB(0);
    storeA(0, 0);
    if (fast0) asyncB(0, 0); else storeB(0, 0);

    int buf = 0;
    for (int it = 0; it < nt; ++it) {
        const bool hasNext = (it + 1 < nt);
        const int  nk = (it + 1) * 32;
        const bool nFast = hasNext && wAsync8 && bFull && (nk + 32 <= K);
        if (hasNext) {
            loadA(nk);                       // A prefetch (regs only)
            if (!nFast) loadB(nk);
        }
        if (WB) asm volatile("s_wait_asynccnt 0x0" ::: "memory");  // own async copies landed
        __syncthreads();   // tile `buf` visible; prior reads of buf^1 already consumed by wmma

        int r = lane & 15, kb = (lane >> 4) * 16;
        v16bf af[2], bfm[2];
        #pragma unroll
        for (int mt = 0; mt < 2; ++mt) {
            int row = wm * 32 + mt * 16 + r;
            union { v16bf v; v8bf h[2]; } u;
            u.h[0] = *(const v8bf*)&As[buf][row][kb];
            u.h[1] = *(const v8bf*)&As[buf][row][kb + 8];
            af[mt] = u.v;
        }
        #pragma unroll
        for (int nt2 = 0; nt2 < 2; ++nt2) {
            int row = wn * 32 + nt2 * 16 + r;
            union { v16bf v; v8bf h[2]; } u;
            u.h[0] = *(const v8bf*)&Bs[buf][row][kb];
            u.h[1] = *(const v8bf*)&Bs[buf][row][kb + 8];
            bfm[nt2] = u.v;
        }
        #pragma unroll
        for (int mt = 0; mt < 2; ++mt)
            #pragma unroll
            for (int nt2 = 0; nt2 < 2; ++nt2)
                acc[mt][nt2] = __builtin_amdgcn_wmma_f32_16x16x32_bf16(
                    false, af[mt], false, bfm[nt2], (short)0, acc[mt][nt2], false, false);

        if (hasNext) {
            storeA(buf ^ 1, nk);
            if (nFast) asyncB(buf ^ 1, nk); else storeB(buf ^ 1, nk);
            buf ^= 1;
        }
    }

    // ---------------- epilogue ----------------
    int cr = lane >> 4, cc = lane & 15;
    if (aFull && bFull) {
        #pragma unroll
        for (int mt = 0; mt < 2; ++mt)
            #pragma unroll
            for (int nt2 = 0; nt2 < 2; ++nt2)
                #pragma unroll
                for (int r8 = 0; r8 < 8; ++r8) {
                    int gr = row0 + wm * 32 + mt * 16 + cr * 8 + r8;
                    int gc = col0 + wn * 32 + nt2 * 16 + cc;
                    float v = acc[mt][nt2][r8] * scale;
                    if (bias) v += bias[gc];
                    if (Rp)   v += Rp[(long long)gr * ldc + gc];
                    if (ACT == 1) v = fmaxf(v, 0.0f);
                    if (ACT == 2) {
                        float u = 0.7978845608028654f * (v + 0.044715f * v * v * v);
                        v = 0.5f * v * (1.0f + tanhf(u));
                    }
                    Cout[(long long)gr * ldc + gc] = v;
                }
    } else {
        #pragma unroll
        for (int mt = 0; mt < 2; ++mt)
            #pragma unroll
            for (int nt2 = 0; nt2 < 2; ++nt2)
                #pragma unroll
                for (int r8 = 0; r8 < 8; ++r8) {
                    int gr = row0 + wm * 32 + mt * 16 + cr * 8 + r8;
                    int gc = col0 + wn * 32 + nt2 * 16 + cc;
                    if (gr < M && gc < Ncols) {
                        float v = acc[mt][nt2][r8] * scale;
                        if (bias) v += bias[gc];
                        if (Rp)   v += Rp[(long long)gr * ldc + gc];
                        if (ACT == 1) v = fmaxf(v, 0.0f);
                        if (ACT == 2) {
                            float u = 0.7978845608028654f * (v + 0.044715f * v * v * v);
                            v = 0.5f * v * (1.0f + tanhf(u));
                        }
                        Cout[(long long)gr * ldc + gc] = v;
                    }
                }
    }
}

// ---- launch wrappers ----
template<int ACT>
static void gemm_w(const float* A, const __bf16* Wt, const float* bias, const float* R,
                   float* Cd, int M, int K, int Nc, int lda, int ldc, float scale, hipStream_t s) {
    dim3 grid((Nc + 63) / 64, (M + 63) / 64, 1), blk(128);
    gemm_k<ACT, 1, 1><<<grid, blk, 0, s>>>(A, (const void*)Wt, bias, R, Cd, M, K, Nc,
                                           lda, /*ldw=*/K, ldc, scale, 0, 0, 0, 0, 0, 0, 1);
}

template<int ACT, int WT>
static void gemm_batched(const float* A, const float* W, const float* bias, const float* R,
                         float* Cd, int M, int K, int Nc, int lda, int ldw, int ldc, float scale,
                         long long aB, long long aH, long long wB, long long wH,
                         long long cB, long long cH, int Hn, int Z, hipStream_t s) {
    dim3 grid((Nc + 63) / 64, (M + 63) / 64, Z), blk(128);
    gemm_k<ACT, WT, 0><<<grid, blk, 0, s>>>(A, (const void*)W, bias, R, Cd, M, K, Nc,
                                            lda, ldw, ldc, scale, aB, aH, wB, wH, cB, cH, Hn);
}

// ============================================================
// Weight convert + transpose: src f32 K x N  ->  dst bf16 N x K
// ============================================================
__global__ void wcvt_k(const float* __restrict__ src, __bf16* __restrict__ dst, int K, int Nc) {
    __shared__ float tile[32][33];
    int bx = blockIdx.x * 32;                 // n base
    int by = blockIdx.y * 32;                 // k base
    int tx = threadIdx.x & 31, ty = threadIdx.x >> 5;   // 32 x 8
    #pragma unroll
    for (int r = 0; r < 32; r += 8) {
        int k = by + ty + r, n = bx + tx;
        tile[ty + r][tx] = (k < K && n < Nc) ? src[(long long)k * Nc + n] : 0.0f;
    }
    __syncthreads();
    #pragma unroll
    for (int r = 0; r < 32; r += 8) {
        int n = bx + ty + r, k = by + tx;
        if (n < Nc && k < K) dst[(long long)n * K + k] = (__bf16)tile[tx][ty + r];
    }
}

// ============================================================
// Elementwise / reduction kernels
// ============================================================
__global__ void layernorm_rows(const float* __restrict__ x, const float* __restrict__ g,
                               const float* __restrict__ bta, float* __restrict__ y) {
    long long row = blockIdx.x;
    x += row * D; y += row * D;
    int t = threadIdx.x;
    float v = x[t];
    __shared__ float red[256];
    red[t] = v; __syncthreads();
    for (int off = 128; off > 0; off >>= 1) { if (t < off) red[t] += red[t + off]; __syncthreads(); }
    float mean = red[0] / D; __syncthreads();
    float d0 = v - mean;
    red[t] = d0 * d0; __syncthreads();
    for (int off = 128; off > 0; off >>= 1) { if (t < off) red[t] += red[t + off]; __syncthreads(); }
    float var = red[0] / D;
    y[t] = g[t] * d0 * rsqrtf(var + 1e-5f) + bta[t];
}

__global__ void softmax_rows(float* __restrict__ x, int cols) {
    long long row = blockIdx.x;
    x += row * cols;
    int t = threadIdx.x;
    __shared__ float red[256];
    float m = -3.4e38f;
    for (int c = t; c < cols; c += 256) m = fmaxf(m, x[c]);
    red[t] = m; __syncthreads();
    for (int off = 128; off > 0; off >>= 1) { if (t < off) red[t] = fmaxf(red[t], red[t + off]); __syncthreads(); }
    m = red[0]; __syncthreads();
    float sum = 0.f;
    for (int c = t; c < cols; c += 256) { float e = expf(x[c] - m); x[c] = e; sum += e; }
    red[t] = sum; __syncthreads();
    for (int off = 128; off > 0; off >>= 1) { if (t < off) red[t] += red[t + off]; __syncthreads(); }
    float inv = 1.0f / red[0];
    for (int c = t; c < cols; c += 256) x[c] *= inv;
}

__global__ void seg_accum_k(const float* __restrict__ pf, const float* __restrict__ pc,
                            const int* __restrict__ sv, float* psum, float* pcnt, float* pcsum) {
    long long p = blockIdx.x;
    int b = (int)(p / N);
    int id = sv[p];
    long long base = (long long)b * S + id;
    atomicAdd(&psum[base * D + threadIdx.x], pf[p * D + threadIdx.x]);
    if (threadIdx.x == 0) atomicAdd(&pcnt[base], 1.0f);
    if (threadIdx.x < 3)  atomicAdd(&pcsum[base * 3 + threadIdx.x], pc[p * 3 + threadIdx.x]);
}

__global__ void seg_fin_k(const float* psum, const float* pcnt, const float* pcsum,
                          float* pmean, float* pcmean) {
    long long bs = blockIdx.x;
    float c = pcnt[bs];
    float inv = (c > 0.f) ? 1.0f / c : 0.0f;
    pmean[bs * D + threadIdx.x] = psum[bs * D + threadIdx.x] * inv;
    if (threadIdx.x < 3) pcmean[bs * 3 + threadIdx.x] = pcsum[bs * 3 + threadIdx.x] * inv;
}

__global__ void consist_k(const float* __restrict__ pf, const int* __restrict__ sv,
                          const float* __restrict__ pooled, float* acc) {
    long long p = blockIdx.x;
    int b = (int)(p / N);
    int id = sv[p];
    float d0 = pf[p * D + threadIdx.x] - pooled[((long long)b * S + id) * D + threadIdx.x];
    __shared__ float red[256];
    red[threadIdx.x] = d0 * d0; __syncthreads();
    for (int off = 128; off > 0; off >>= 1) { if (threadIdx.x < off) red[threadIdx.x] += red[threadIdx.x + off]; __syncthreads(); }
    if (threadIdx.x == 0) atomicAdd(acc, red[0]);
}

__global__ void scale1_k(float* p, float s) { if (threadIdx.x == 0 && blockIdx.x == 0) *p *= s; }

__global__ void build_view_k(const float* cls, const float* clsp, const float* ivt,
                             const float* vpe, float* out) {
    int b = blockIdx.x / TV, t = blockIdx.x % TV, d0 = threadIdx.x;
    float v = (t < C) ? cls[t * D + d0] + clsp[t * D + d0]
                      : ivt[((long long)b * V + (t - C)) * D + d0] + vpe[(t - C) * D + d0];
    out[((long long)b * TV + t) * D + d0] = v;
}

__global__ void build_vox_k(const float* cls, const float* clsp, const float* pmean,
                            const float* pos, float* out) {
    int b = blockIdx.x / TX, t = blockIdx.x % TX, d0 = threadIdx.x;
    float v;
    if (t < C) v = cls[t * D + d0] + clsp[t * D + d0];
    else {
        long long idx = ((long long)b * S + (t - C)) * D + d0;
        v = pmean[idx] + pos[idx];
    }
    out[((long long)b * TX + t) * D + d0] = v;
}

__global__ void relu_rows_k(const float* x, float* y) {
    int b = blockIdx.x / S, s0 = blockIdx.x % S;
    y[((long long)b * S + s0) * D + threadIdx.x] =
        fmaxf(0.f, x[((long long)b * TX + C + s0) * D + threadIdx.x]);
}

__global__ void mean_rows_D_k(const float* x, float* out, int T) {
    int b = blockIdx.x / C, c = blockIdx.x % C;
    const float* r = x + ((long long)b * T + c) * D;
    __shared__ float red[256];
    red[threadIdx.x] = r[threadIdx.x]; __syncthreads();
    for (int off = 128; off > 0; off >>= 1) { if (threadIdx.x < off) red[threadIdx.x] += red[threadIdx.x + off]; __syncthreads(); }
    if (threadIdx.x == 0) out[b * C + c] = red[0] / D;
}

__global__ void cross_mean_k(const float* vx, const float* xx, float* out) {
    int b = blockIdx.x / C, c = blockIdx.x % C;
    float a = vx[((long long)b * TV + c) * D + threadIdx.x];
    float d0 = xx[((long long)b * TX + c) * D + threadIdx.x];
    __shared__ float red[256];
    red[threadIdx.x] = 0.5f * (a + d0); __syncthreads();
    for (int off = 128; off > 0; off >>= 1) { if (threadIdx.x < off) red[threadIdx.x] += red[threadIdx.x + off]; __syncthreads(); }
    if (threadIdx.x == 0) out[b * C + c] = red[0] / D;
}

__global__ void mean_S_k(const float* vc, float* out) {
    int b = blockIdx.x / C, c = blockIdx.x % C;
    float s = 0.f;
    for (int i = threadIdx.x; i < S; i += 256) s += vc[((long long)b * S + i) * C + c];
    __shared__ float red[256];
    red[threadIdx.x] = s; __syncthreads();
    for (int off = 128; off > 0; off >>= 1) { if (threadIdx.x < off) red[threadIdx.x] += red[threadIdx.x + off]; __syncthreads(); }
    if (threadIdx.x == 0) out[b * C + c] = red[0] / S;
}

__global__ void mean_V_k(const float* t, float* out) {
    int i = threadIdx.x;
    if (i >= B * C) return;
    int b = i / C, c = i % C;
    float s = 0.f;
    for (int v = 0; v < V; ++v) s += t[((long long)b * V + v) * C + c];
    out[i] = s / V;
}

__global__ void simloss_k(const float* __restrict__ p, float* acc, int Tq, int Tk) {
    int b = blockIdx.x;
    __shared__ float sim[C][C];
    __shared__ float part[C];
    for (int idx = threadIdx.x; idx < C * C; idx += blockDim.x) {
        int i = idx / C, j = idx % C;
        float s = 0.f;
        for (int h = 0; h < H; ++h) s += p[(((long long)b * H + h) * Tq + i) * Tk + j];
        sim[i][j] = s / H;
    }
    __syncthreads();
    if (threadIdx.x < C) {
        int i = threadIdx.x;
        float mr = -3.4e38f, mc = -3.4e38f;
        for (int j = 0; j < C; ++j) { mr = fmaxf(mr, sim[i][j]); mc = fmaxf(mc, sim[j][i]); }
        float sr = 0.f, sc = 0.f;
        for (int j = 0; j < C; ++j) { sr += expf(sim[i][j] - mr); sc += expf(sim[j][i] - mc); }
        float lser = mr + logf(sr), lsec = mc + logf(sc);
        part[i] = (lser - sim[i][i]) + (lsec - sim[i][i]);
    }
    __syncthreads();
    if (threadIdx.x == 0) {
        float t = 0.f;
        for (int i = 0; i < C; ++i) t += part[i];
        t = 0.5f * t / C;                       // (lv+lx)*0.5 for this batch
        atomicAdd(acc, t / (DEPTH * B));        // mean over layers and batch
    }
}

// ============================================================
// Host orchestration
// ============================================================
extern "C" void kernel_launch(void* const* d_in, const int* in_sizes, int n_in,
                              void* d_out, int out_size, void* d_ws, size_t ws_size,
                              hipStream_t stream) {
    (void)in_sizes; (void)n_in; (void)out_size; (void)ws_size;
    typedef long long ll;

    const float* point_feat  = (const float*)d_in[0];
    const float* point_coord = (const float*)d_in[1];
    const float* image_view  = (const float*)d_in[2];
    const int*   supervoxel  = (const int*)  d_in[3];
    const float* cls_token   = (const float*)d_in[4];
    const float* cls_pos     = (const float*)d_in[5];
    const float* view_pos    = (const float*)d_in[6];
    const float* pos_w1 = (const float*)d_in[7];  const float* pos_b1 = (const float*)d_in[8];
    const float* pos_w2 = (const float*)d_in[9];  const float* pos_b2 = (const float*)d_in[10];
    const float* pos_w3 = (const float*)d_in[11]; const float* pos_b3 = (const float*)d_in[12];
    const float* enc_qkv_w  = (const float*)d_in[13]; const float* enc_qkv_b  = (const float*)d_in[14];
    const float* enc_proj_w = (const float*)d_in[15]; const float* enc_proj_b = (const float*)d_in[16];
    const float* enc_ln1_g  = (const float*)d_in[17]; const float* enc_ln1_b  = (const float*)d_in[18];
    const float* enc_ln2_g  = (const float*)d_in[19]; const float* enc_ln2_b  = (const float*)d_in[20];
    const float* enc_mlp_w1 = (const float*)d_in[21]; const float* enc_mlp_b1 = (const float*)d_in[22];
    const float* enc_mlp_w2 = (const float*)d_in[23]; const float* enc_mlp_b2 = (const float*)d_in[24];
    const float* dec_q_w   = (const float*)d_in[25]; const float* dec_q_b   = (const float*)d_in[26];
    const float* dec_kv_w  = (const float*)d_in[27]; const float* dec_kv_b  = (const float*)d_in[28];
    const float* dec_pj_w  = (const float*)d_in[29]; const float* dec_pj_b  = (const float*)d_in[30];
    const float* dec_lnq_g = (const float*)d_in[31]; const float* dec_lnq_b = (const float*)d_in[32];
    const float* dec_lnk_g = (const float*)d_in[33]; const float* dec_lnk_b = (const float*)d_in[34];
    const float* dec_ln2_g = (const float*)d_in[35]; const float* dec_ln2_b = (const float*)d_in[36];
    const float* dec_mw1 = (const float*)d_in[37]; const float* dec_mb1 = (const float*)d_in[38];
    const float* dec_mw2 = (const float*)d_in[39]; const float* dec_mb2 = (const float*)d_in[40];
    const float* head2d_w = (const float*)d_in[41]; const float* head2d_b = (const float*)d_in[42];
    const float* head3d_w = (const float*)d_in[43]; const float* head3d_b = (const float*)d_in[44];

    float* out = (float*)d_out;
    float* Wk  = (float*)d_ws;

    // ---------------- f32 workspace layout ----------------
    size_t o = 0;
    auto alloc = [&](size_t n) { size_t r = o; o += n; return r; };
    float* pooled_sum  = Wk + alloc((size_t)B * S * D);
    float* pooled_cnt  = Wk + alloc((size_t)B * S);
    float* pcoord_sum  = Wk + alloc((size_t)B * S * 3);
    float* pooled_feat = Wk + alloc((size_t)B * S * D);
    float* pcoord_mean = Wk + alloc((size_t)B * S * 3 + 4);
    float* posh1       = Wk + alloc((size_t)B * S * D);
    float* posh2       = Wk + alloc((size_t)B * S * D);
    float* posout      = Wk + alloc((size_t)B * S * D);
    float* view_x      = Wk + alloc((size_t)B * TV * D);
    float* vox_x       = Wk + alloc((size_t)B * TX * D);
    float* lnA         = Wk + alloc((size_t)B * TX * D);   // xk / encoder h / vox mlp h
    float* lnB         = Wk + alloc((size_t)B * TX * D);   // xq
    float* lnDq        = Wk + alloc((size_t)B * TV * D);   // vq / view mlp h
    float* lnE         = Wk + alloc((size_t)B * TV * D);   // vk
    float* qkvbuf      = Wk + alloc((size_t)B * TX * 3 * D);
    float* scores      = Wk + alloc((size_t)B * H * TX * TX);
    float* obuf        = Wk + alloc((size_t)B * TX * D);
    float* mlpbuf      = Wk + alloc((size_t)B * TX * MH);
    float* vcamtmp     = Wk + alloc((size_t)B * V * C);
    float* rbuf        = Wk + alloc((size_t)B * S * D);
    float* qbuf  = qkvbuf;
    float* kvbuf = qkvbuf + (ll)B * TX * D;

    // ---------------- bf16 (transposed) weight arena ----------------
    o = (o + 3) & ~(size_t)3;
    __bf16* bw = (__bf16*)(Wk + o);
    size_t ob = 0;
    auto balloc = [&](size_t n) { size_t r = ob; ob += (n + 7) & ~(size_t)7; return r; };
    __bf16* b_pos1  = bw + balloc((size_t)3 * D);
    __bf16* b_pos2  = bw + balloc((size_t)D * D);
    __bf16* b_pos3  = bw + balloc((size_t)D * D);
    __bf16* b_eqkv  = bw + balloc((size_t)8 * D * 3 * D);
    __bf16* b_eproj = bw + balloc((size_t)8 * D * D);
    __bf16* b_emw1  = bw + balloc((size_t)8 * D * MH);
    __bf16* b_emw2  = bw + balloc((size_t)8 * MH * D);
    __bf16* b_dq    = bw + balloc((size_t)8 * D * D);
    __bf16* b_dkv   = bw + balloc((size_t)8 * D * 2 * D);
    __bf16* b_dpj   = bw + balloc((size_t)8 * D * D);
    __bf16* b_dmw1  = bw + balloc((size_t)8 * D * MH);
    __bf16* b_dmw2  = bw + balloc((size_t)8 * MH * D);
    __bf16* b_h2d   = bw + balloc((size_t)D * C);
    __bf16* b_h3d   = bw + balloc((size_t)D * C);

    auto wconv = [&](const float* src, __bf16* dst, int K, int Nc) {
        dim3 g((Nc + 31) / 32, (K + 31) / 32);
        wcvt_k<<<g, 256, 0, stream>>>(src, dst, K, Nc);
    };
    wconv(pos_w1, b_pos1, 3, D);
    wconv(pos_w2, b_pos2, D, D);
    wconv(pos_w3, b_pos3, D, D);
    for (int i = 0; i < 2 * DEPTH; ++i) {
        wconv(enc_qkv_w  + (ll)i * D * 3 * D, b_eqkv  + (ll)i * D * 3 * D, D, 3 * D);
        wconv(enc_proj_w + (ll)i * D * D,     b_eproj + (ll)i * D * D,     D, D);
        wconv(enc_mlp_w1 + (ll)i * D * MH,    b_emw1  + (ll)i * D * MH,    D, MH);
        wconv(enc_mlp_w2 + (ll)i * MH * D,    b_emw2  + (ll)i * MH * D,    MH, D);
        wconv(dec_q_w    + (ll)i * D * D,     b_dq    + (ll)i * D * D,     D, D);
        wconv(dec_kv_w   + (ll)i * D * 2 * D, b_dkv   + (ll)i * D * 2 * D, D, 2 * D);
        wconv(dec_pj_w   + (ll)i * D * D,     b_dpj   + (ll)i * D * D,     D, D);
        wconv(dec_mw1    + (ll)i * D * MH,    b_dmw1  + (ll)i * D * MH,    D, MH);
        wconv(dec_mw2    + (ll)i * MH * D,    b_dmw2  + (ll)i * MH * D,    MH, D);
    }
    wconv(head2d_w, b_h2d, D, C);
    wconv(head3d_w, b_h3d, D, C);

    // ---------------- zero accumulators ----------------
    hipMemsetAsync(pooled_sum, 0, ((size_t)B * S * D + B * S + B * S * 3) * sizeof(float), stream);
    hipMemsetAsync(out + 400, 0, 2 * sizeof(float), stream);

    // ---------------- segment pooling (memory-bound: 2 passes over 245MB) ----
    seg_accum_k<<<B * N, 256, 0, stream>>>(point_feat, point_coord, supervoxel,
                                           pooled_sum, pooled_cnt, pcoord_sum);
    seg_fin_k<<<B * S, 256, 0, stream>>>(pooled_sum, pooled_cnt, pcoord_sum,
                                         pooled_feat, pcoord_mean);
    consist_k<<<B * N, 256, 0, stream>>>(point_feat, supervoxel, pooled_feat, out + 400);
    scale1_k<<<1, 32, 0, stream>>>(out + 400, 1.0f / ((float)B * N * D));

    // ---------------- position MLP ----------------
    gemm_w<1>(pcoord_mean, b_pos1, pos_b1, nullptr, posh1, B * S, 3, D, 3, D, 1.f, stream);
    gemm_w<1>(posh1, b_pos2, pos_b2, nullptr, posh2, B * S, D, D, D, D, 1.f, stream);
    gemm_w<0>(posh2, b_pos3, pos_b3, nullptr, posout, B * S, D, D, D, D, 1.f, stream);

    // ---------------- encoder (self-attn, WMMA GEMMs) ----------------
    auto run_encoder = [&](float* x, int T, int e) {
        int M = B * T;
        for (int d = 0; d < DEPTH; ++d) {
            ll wi = (ll)(e * DEPTH + d);
            layernorm_rows<<<M, 256, 0, stream>>>(x, enc_ln1_g + wi * D, enc_ln1_b + wi * D, lnA);
            gemm_w<0>(lnA, b_eqkv + wi * D * 3 * D, enc_qkv_b + wi * 3 * D, nullptr,
                      qkvbuf, M, D, 3 * D, D, 3 * D, 1.f, stream);
            // scores = SCALE * Q @ K^T  (batched over B*H)
            gemm_batched<0, 1>(qkvbuf, qkvbuf + D, nullptr, nullptr, scores,
                               T, DH, T, 3 * D, 3 * D, T, SCALE_ATT,
                               (ll)T * 3 * D, DH, (ll)T * 3 * D, DH, (ll)H * T * T, (ll)T * T,
                               H, B * H, stream);
            softmax_rows<<<B * H * T, 256, 0, stream>>>(scores, T);
            // O = P @ V
            gemm_batched<0, 0>(scores, qkvbuf + 2 * D, nullptr, nullptr, obuf,
                               T, T, DH, T, 3 * D, D, 1.f,
                               (ll)H * T * T, (ll)T * T, (ll)T * 3 * D, DH, (ll)T * D, DH,
                               H, B * H, stream);
            gemm_w<0>(obuf, b_eproj + wi * D * D, enc_proj_b + wi * D, x, x,
                      M, D, D, D, D, 1.f, stream);
            layernorm_rows<<<M, 256, 0, stream>>>(x, enc_ln2_g + wi * D, enc_ln2_b + wi * D, lnA);
            gemm_w<2>(lnA, b_emw1 + wi * D * MH, enc_mlp_b1 + wi * MH, nullptr,
                      mlpbuf, M, D, MH, D, MH, 1.f, stream);
            gemm_w<0>(mlpbuf, b_emw2 + wi * MH * D, enc_mlp_b2 + wi * D, x, x,
                      M, MH, D, MH, D, 1.f, stream);
        }
    };

    build_view_k<<<B * TV, 256, 0, stream>>>(cls_token, cls_pos, image_view, view_pos, view_x);
    run_encoder(view_x, TV, 0);
    mean_rows_D_k<<<B * C, 256, 0, stream>>>(view_x, out + 80, TV);        // mct_logits_2d

    build_vox_k<<<B * TX, 256, 0, stream>>>(cls_token, cls_pos, pooled_feat, posout, vox_x);
    run_encoder(vox_x, TX, 1);
    mean_rows_D_k<<<B * C, 256, 0, stream>>>(vox_x, out + 160, TX);        // mct_logits_3d

    // heads on encoder outputs (before decoder mutates vox_x/view_x)
    relu_rows_k<<<B * S, 256, 0, stream>>>(vox_x, rbuf);
    gemm_w<0>(rbuf, b_h3d, head3d_b, nullptr, out + 402, B * S, D, C, D, C, 1.f, stream);
    mean_S_k<<<B * C, 256, 0, stream>>>(out + 402, out + 320);             // cam_logits_3d
    gemm_w<0>(image_view, b_h2d, head2d_b, nullptr, vcamtmp, B * V, D, C, D, C, 1.f, stream);
    mean_V_k<<<1, 128, 0, stream>>>(vcamtmp, out + 240);                   // cam_logits_2d

    // ---------------- cross-attention decoder ----------------
    auto run_cross_mha = [&](const float* qin, const float* kvin, int Tq, int Tk,
                             int sside, int d, float* xout, bool sim) {
        ll wi = (ll)(sside * DEPTH + d);
        gemm_w<0>(qin, b_dq + wi * D * D, dec_q_b + wi * D, nullptr, qbuf,
                  B * Tq, D, D, D, D, 1.f, stream);
        gemm_w<0>(kvin, b_dkv + wi * D * 2 * D, dec_kv_b + wi * 2 * D, nullptr, kvbuf,
                  B * Tk, D, 2 * D, D, 2 * D, 1.f, stream);
        gemm_batched<0, 1>(qbuf, kvbuf, nullptr, nullptr, scores,
                           Tq, DH, Tk, D, 2 * D, Tk, SCALE_ATT,
                           (ll)Tq * D, DH, (ll)Tk * 2 * D, DH, (ll)H * Tq * Tk, (ll)Tq * Tk,
                           H, B * H, stream);
        softmax_rows<<<B * H * Tq, 256, 0, stream>>>(scores, Tk);
        if (sim) simloss_k<<<B, 256, 0, stream>>>(scores, out + 401, Tq, Tk);
        gemm_batched<0, 0>(scores, kvbuf + D, nullptr, nullptr, obuf,
                           Tq, Tk, DH, Tk, 2 * D, D, 1.f,
                           (ll)H * Tq * Tk, (ll)Tq * Tk, (ll)Tk * 2 * D, DH, (ll)Tq * D, DH,
                           H, B * H, stream);
        gemm_w<0>(obuf, b_dpj + wi * D * D, dec_pj_b + wi * D, xout, xout,
                  B * Tq, D, D, D, D, 1.f, stream);
    };

    for (int d = 0; d < DEPTH; ++d) {
        ll w0 = (ll)(0 * DEPTH + d) * D, w1 = (ll)(1 * DEPTH + d) * D;
        layernorm_rows<<<B * TV, 256, 0, stream>>>(view_x, dec_lnq_g + w0, dec_lnq_b + w0, lnDq); // vq
        layernorm_rows<<<B * TX, 256, 0, stream>>>(vox_x,  dec_lnk_g + w0, dec_lnk_b + w0, lnA);  // xk
        layernorm_rows<<<B * TX, 256, 0, stream>>>(vox_x,  dec_lnq_g + w1, dec_lnq_b + w1, lnB);  // xq
        layernorm_rows<<<B * TV, 256, 0, stream>>>(view_x, dec_lnk_g + w1, dec_lnk_b + w1, lnE);  // vk
        run_cross_mha(lnDq, lnA, TV, TX, 0, d, view_x, true);
        run_cross_mha(lnB,  lnE, TX, TV, 1, d, vox_x, false);
        // MLPs
        layernorm_rows<<<B * TV, 256, 0, stream>>>(view_x, dec_ln2_g + w0, dec_ln2_b + w0, lnDq);
        gemm_w<2>(lnDq, b_dmw1 + (ll)(0 * DEPTH + d) * D * MH, dec_mb1 + (ll)(0 * DEPTH + d) * MH,
                  nullptr, mlpbuf, B * TV, D, MH, D, MH, 1.f, stream);
        gemm_w<0>(mlpbuf, b_dmw2 + (ll)(0 * DEPTH + d) * MH * D, dec_mb2 + (ll)(0 * DEPTH + d) * D,
                  view_x, view_x, B * TV, MH, D, MH, D, 1.f, stream);
        layernorm_rows<<<B * TX, 256, 0, stream>>>(vox_x, dec_ln2_g + w1, dec_ln2_b + w1, lnA);
        gemm_w<2>(lnA, b_dmw1 + (ll)(1 * DEPTH + d) * D * MH, dec_mb1 + (ll)(1 * DEPTH + d) * MH,
                  nullptr, mlpbuf, B * TX, D, MH, D, MH, 1.f, stream);
        gemm_w<0>(mlpbuf, b_dmw2 + (ll)(1 * DEPTH + d) * MH * D, dec_mb2 + (ll)(1 * DEPTH + d) * D,
                  vox_x, vox_x, B * TX, MH, D, MH, D, 1.f, stream);
    }

    cross_mean_k<<<B * C, 256, 0, stream>>>(view_x, vox_x, out + 0);       // cross_mct_logits
}
